// GCN_46643344834644
// MI455X (gfx1250) — matrix operand order repaired and checked by reading, not compiled, not run on hardware
//
#include <hip/hip_runtime.h>
#include <hip/hip_bf16.h>

typedef _Float16 f16;
typedef _Float16 v16h  __attribute__((ext_vector_type(16)));
typedef _Float16 half8 __attribute__((ext_vector_type(8)));
typedef float    v8f   __attribute__((ext_vector_type(8)));

#define NB   128      // nodes
#define EE   2048     // edges
#define BB   8
#define TT   32
#define ROWS 32768    // B*T*N
#define BN_ROWS 1024  // B*N

__device__ __forceinline__ v16h cat16(half8 lo, half8 hi) {
  return __builtin_shufflevector(lo, hi, 0,1,2,3,4,5,6,7,8,9,10,11,12,13,14,15);
}

// ---------------------------------------------------------------- graph setup
// Deterministic: per-destination CSR built by an ordered scan (no atomics).
__global__ void k_graph(const int* __restrict__ ei, const float* __restrict__ ew,
                        int* __restrict__ csrSrc, float* __restrict__ csrW,
                        int* __restrict__ csrCnt)
{
  __shared__ float dinv[NB];
  int n = threadIdx.x;
  float deg = 0.f;
  for (int e = 0; e < EE; ++e) if (ei[2*e] == n) deg += ew[e];
  dinv[n] = (deg > 0.f) ? rsqrtf(deg) : 0.f;
  __syncthreads();
  int cnt = 0;
  for (int e = 0; e < EE; ++e) {
    if (ei[2*e + 1] == n) {
      int s = ei[2*e];
      csrSrc[n*EE + cnt] = s;
      csrW  [n*EE + cnt] = -dinv[s] * ew[e] * dinv[n];   // L_hat edge value
      ++cnt;
    }
  }
  csrCnt[n] = cnt;
}

// ------------------------------------------------------------- generic packer
// Packs a row-major f32 (Ksrc x Nsrc) matrix into WMMA-ready f16 B tiles of a
// padded (Kp x Np) matrix.  Tile (kt,nt) -> 512 halves; lane l, elem e holds
// B[kt*32 + (l>=16)*16 + e][nt*16 + (l&15)].
__global__ void k_pack_b(const float* __restrict__ src, f16* __restrict__ dst,
                         int Ksrc, int Nsrc, int Kp, int Np,
                         long srcB, long dstB)
{
  long total = (long)(Kp >> 5) * (Np >> 4) * 512;
  long p = (long)blockIdx.x * 256 + threadIdx.x;
  if (p >= total) return;
  const float* s = src + (long)blockIdx.y * srcB;
  f16* d = dst + (long)blockIdx.y * dstB;
  long tile = p >> 9; int r = (int)(p & 511);
  int lane = r >> 4, e = r & 15;
  int nt16 = Np >> 4;
  int kt = (int)(tile / nt16), nt = (int)(tile % nt16);
  int k = kt*32 + (lane >> 4)*16 + e;
  int n = nt*16 + (lane & 15);
  float v = (k < Ksrc && n < Nsrc) ? s[(long)k * Nsrc + n] : 0.f;
  d[p] = (f16)v;
}

// Pack the per-layer concatenated LSTM weight  Wcat (1536 x 1024):
// rows: [cxW k=0..2 | chW k=0..2] each 256; cols: gate j*256 + h.
__global__ void k_pack_wcat(const float* __restrict__ cxW,
                            const float* __restrict__ chW, f16* __restrict__ dst)
{
  const long total = 48L * 64 * 512;          // (1536/32)*(1024/16)*512
  long p = (long)blockIdx.x * 256 + threadIdx.x;
  if (p >= total) return;
  int l = blockIdx.y;
  long tile = p >> 9; int r = (int)(p & 511);
  int lane = r >> 4, e = r & 15;
  int kt = (int)(tile / 64), nt = (int)(tile % 64);
  int k = kt*32 + (lane >> 4)*16 + e;         // 0..1535
  int n = nt*16 + (lane & 15);                // 0..1023
  int j = n >> 8, h = n & 255;
  int s = k >> 8, kk = k & 255;
  float v;
  if (s < 3) v = cxW[(((long)(l*4 + j)*3 + s     )*256 + kk)*256 + h];
  else       v = chW[(((long)(l*4 + j)*3 + (s-3))*256 + kk)*256 + h];
  dst[(long)l * total + p] = (f16)v;
}

__global__ void k_biascat(const float* __restrict__ cxb, const float* __restrict__ chb,
                          const float* __restrict__ gb, float* __restrict__ out)
{
  int i = blockIdx.x * 256 + threadIdx.x;
  if (i >= 2048) return;
  out[i] = cxb[i] + chb[i] + gb[i];           // all are (L,4,HID) flat
}

// ------------------------------------------------------------------ WMMA GEMM
// C(MxN,f32) = A(MxK,f16,row-major) @ Bpacked(KxN,f16 tiles) [+ bias[col]]
// wave tile = 32x32 (2x2 of 16x16x32 WMMA), K-step 32.  Batched via grid.y.
__global__ __launch_bounds__(256)
void k_wmma_gemm(const f16* __restrict__ A, const f16* __restrict__ Bp,
                 float* __restrict__ C, const float* __restrict__ bias,
                 int M, int N, int K, int lda, int ldc,
                 long sAb, long sBb, long sCb, long sBiasB)
{
  const int wid  = blockIdx.x * 8 + (threadIdx.x >> 5);
  const int lane = threadIdx.x & 31;
  const int tilesN = N >> 5;
  const int tm = wid / tilesN;
  const int tn = wid - tm * tilesN;
  if (tm * 32 >= M) return;

  const int z = blockIdx.y;
  A  += (long)z * sAb;
  Bp += (long)z * sBb;
  C  += (long)z * sCb;
  const float* bz = bias ? (bias + (long)z * sBiasB) : nullptr;

  const int rsel = lane & 15;      // A row / C col selector
  const int hsel = lane >> 4;      // K-half / C row-half selector
  const int nt16 = N >> 4;

  v8f acc00 = {}, acc01 = {}, acc10 = {}, acc11 = {};

  for (int k0 = 0; k0 < K; k0 += 32) {
    const f16* p0 = A + (long)(tm*32 + rsel) * lda + k0 + hsel*16;
    half8 a0lo = *(const half8*)p0;
    half8 a0hi = *(const half8*)(p0 + 8);
    const f16* p1 = p0 + (long)16 * lda;
    half8 a1lo = *(const half8*)p1;
    half8 a1hi = *(const half8*)(p1 + 8);
    v16h a0 = cat16(a0lo, a0hi);
    v16h a1 = cat16(a1lo, a1hi);

    long tb = (long)(k0 >> 5) * nt16 + (tn << 1);
    const f16* q0 = Bp + (tb << 9) + lane*16;
    half8 b0lo = *(const half8*)q0;
    half8 b0hi = *(const half8*)(q0 + 8);
    const f16* q1 = q0 + 512;
    half8 b1lo = *(const half8*)q1;
    half8 b1hi = *(const half8*)(q1 + 8);
    v16h b0 = cat16(b0lo, b0hi);
    v16h b1 = cat16(b1lo, b1hi);

    acc00 = __builtin_amdgcn_wmma_f32_16x16x32_f16(false, a0, false, b0, (short)0, acc00, false, false);
    acc01 = __builtin_amdgcn_wmma_f32_16x16x32_f16(false, a0, false, b1, (short)0, acc01, false, false);
    acc10 = __builtin_amdgcn_wmma_f32_16x16x32_f16(false, a1, false, b0, (short)0, acc10, false, false);
    acc11 = __builtin_amdgcn_wmma_f32_16x16x32_f16(false, a1, false, b1, (short)0, acc11, false, false);
  }

  const int col0 = tn*32 + rsel;
  const int col1 = col0 + 16;
  const float bv0 = bz ? bz[col0] : 0.f;
  const float bv1 = bz ? bz[col1] : 0.f;
#pragma unroll
  for (int r = 0; r < 8; ++r) {
    int row0 = tm*32 + hsel*8 + r;
    int row1 = row0 + 16;
    C[(long)row0*ldc + col0] = acc00[r] + bv0;
    C[(long)row0*ldc + col1] = acc01[r] + bv1;
    C[(long)row1*ldc + col0] = acc10[r] + bv0;
    C[(long)row1*ldc + col1] = acc11[r] + bv1;
  }
}

// ------------------------------------------------------------ small kernels
__global__ void k_prep_x(const float* __restrict__ in, f16* __restrict__ Xpad)
{
  long i = (long)blockIdx.x * 256 + threadIdx.x;   // ROWS*32
  if (i >= (long)ROWS * 32) return;
  int f = (int)(i & 31);
  long r = i >> 5;
  int n = (int)(r & 127);
  long q = r >> 7;                                  // b*T+t
  float v = (f < 16) ? in[(q*16 + f)*128 + n] : 0.f;
  Xpad[i] = (f16)v;
}

__global__ void k_colstats(const float* __restrict__ Y, int Mrows, int ld,
                           const float* __restrict__ g, const float* __restrict__ bta,
                           float* __restrict__ scale, float* __restrict__ shift)
{
  __shared__ float s1[256], s2[256];
  int c = blockIdx.x, t = threadIdx.x;
  float s = 0.f, ss = 0.f;
  for (int r = t; r < Mrows; r += 256) {
    float v = Y[(long)r*ld + c]; s += v; ss += v*v;
  }
  s1[t] = s; s2[t] = ss; __syncthreads();
  for (int o = 128; o; o >>= 1) {
    if (t < o) { s1[t] += s1[t+o]; s2[t] += s2[t+o]; }
    __syncthreads();
  }
  if (t == 0) {
    float m  = s1[0] / Mrows;
    float vv = s2[0] / Mrows - m*m;
    float rs = rsqrtf(vv + 1e-5f);
    float sc = rs * g[c];
    scale[c] = sc;
    shift[c] = bta[c] - m * sc;
  }
}

__global__ void k_bnact(const float* __restrict__ Y, f16* __restrict__ H,
                        const float* __restrict__ scale, const float* __restrict__ shift,
                        int cols, long total, int leaky)
{
  long i = (long)blockIdx.x * 256 + threadIdx.x;
  if (i >= total) return;
  int c = (int)(i % cols);
  float v = Y[i]*scale[c] + shift[c];
  if (leaky && v < 0.f) v *= 0.01f;
  H[i] = (f16)v;
}

// BN of the 8-wide inter + reorder into node-major padded (128,256,32) f16 A.
__global__ void k_bn3_reorder(const float* __restrict__ Y, f16* __restrict__ nodeA,
                              const float* __restrict__ scale, const float* __restrict__ shift)
{
  long i = (long)blockIdx.x * 256 + threadIdx.x;   // 128*256*32
  if (i >= 128L*256*32) return;
  int f = (int)(i & 31);
  long rem = i >> 5;
  int q = (int)(rem & 255);
  int n = (int)(rem >> 8);
  float v = 0.f;
  if (f < 8) { float y = Y[((long)q*128 + n)*32 + f]; v = y*scale[f] + shift[f]; }
  nodeA[((long)n*256 + q)*32 + f] = (f16)v;
}

__global__ void k_cvt(const float* __restrict__ src, f16* __restrict__ dst,
                      long total, int leaky)
{
  long i = (long)blockIdx.x * 256 + threadIdx.x;
  if (i >= total) return;
  float v = src[i];
  if (leaky && v < 0.f) v *= 0.01f;
  dst[i] = (f16)v;
}

// node-major emb f32 (128,256,256) -> time-major f16 (T, B*N, 256)
__global__ void k_emb(const float* __restrict__ Y, f16* __restrict__ embT)
{
  long i = (long)blockIdx.x * 256 + threadIdx.x;   // 128*256*256
  if (i >= 128L*256*256) return;
  int c = (int)(i & 255);
  long rem = i >> 8;
  int q = (int)(rem & 255);
  int n = (int)(rem >> 8);
  int b = q >> 5, t = q & 31;
  embT[(((long)t*1024) + b*128 + n)*256 + c] = (f16)Y[i];
}

__global__ void k_zero(unsigned int* __restrict__ p, long words)
{
  long i = (long)blockIdx.x * 256 + threadIdx.x;
  if (i < words) p[i] = 0u;
}

__global__ void k_copy16(f16* __restrict__ dst, long dld,
                         const f16* __restrict__ src, long sld)
{
  int row = blockIdx.x, c = threadIdx.x;
  dst[(long)row*dld + c] = src[(long)row*sld + c];
}

// Graph propagation inside the 1024x1536 concat buffer Acat.
// grid.y: 0 = x-chain (cols 0..), 1 = h-chain (cols 768..)
// phase 0: out(+256) = prop(in(+0));  phase 1: out(+512) = 2*prop(in(+256)) - in(+0)
__global__ void k_prop(f16* __restrict__ Acat, const int* __restrict__ csrSrc,
                       const float* __restrict__ csrW, const int* __restrict__ csrCnt,
                       int phase)
{
  int row = blockIdx.x;           // b*128+n
  int b = row >> 7, n = row & 127;
  int c = threadIdx.x;
  int colBase = blockIdx.y * 768;
  const f16* in = Acat + colBase + (phase ? 256 : 0);
  f16* out      = Acat + colBase + (phase ? 512 : 256);
  int cnt = csrCnt[n];
  const int*   sl = csrSrc + n*EE;
  const float* wl = csrW   + n*EE;
  float acc = 0.f;
  for (int i = 0; i < cnt; ++i) {
    int s = sl[i];
    acc += wl[i] * (float)in[((long)(b*128 + s))*1536 + c];
  }
  if (phase) acc = 2.f*acc - (float)Acat[(long)row*1536 + colBase + c];
  out[(long)row*1536 + c] = (f16)acc;
}

__global__ void k_gates(const float* __restrict__ G, const float* __restrict__ wc,
                        float* __restrict__ Cs, f16* __restrict__ Hs)
{
  int row = blockIdx.x, h = threadIdx.x;
  const float* g = G + (long)row*1024;
  float c = Cs[(long)row*256 + h];
  float I  = 1.f/(1.f + expf(-(g[h      ] + wc[h      ]*c)));
  float F  = 1.f/(1.f + expf(-(g[256 + h] + wc[256 + h]*c)));
  float Cn = F*c + I*tanhf(g[512 + h]);
  float O  = 1.f/(1.f + expf(-(g[768 + h] + wc[512 + h]*Cn)));
  Cs[(long)row*256 + h] = Cn;
  Hs[(long)row*256 + h] = (f16)(O*tanhf(Cn));
}

__global__ void k_out(const f16* __restrict__ H, const float* __restrict__ linW,
                      const float* __restrict__ linb, float* __restrict__ out, int t)
{
  __shared__ float red[256];
  int row = blockIdx.x, c = threadIdx.x;
  red[c] = (float)H[(long)row*256 + c] * linW[c];
  __syncthreads();
  for (int o = 128; o; o >>= 1) { if (c < o) red[c] += red[c+o]; __syncthreads(); }
  if (c == 0) {
    int b = row >> 7, n = row & 127;
    out[(long)b*4096 + t*128 + n] = red[0] + linb[0];
  }
}

// =========================================================================
extern "C" void kernel_launch(void* const* d_in, const int* in_sizes, int n_in,
                              void* d_out, int out_size, void* d_ws, size_t ws_size,
                              hipStream_t stream)
{
  const float* input = (const float*)d_in[0];
  const int*   e_i   = (const int*)  d_in[1];
  const float* e_a   = (const float*)d_in[2];
  const float* W1    = (const float*)d_in[3];
  const float* bn1g  = (const float*)d_in[5];
  const float* bn1b  = (const float*)d_in[6];
  const float* W2    = (const float*)d_in[7];
  const float* bn2g  = (const float*)d_in[9];
  const float* bn2b  = (const float*)d_in[10];
  const float* W3    = (const float*)d_in[11];
  const float* bn3g  = (const float*)d_in[13];
  const float* bn3b  = (const float*)d_in[14];
  const float* nW1   = (const float*)d_in[15];
  const float* nb1   = (const float*)d_in[16];
  const float* nW2   = (const float*)d_in[17];
  const float* nb2   = (const float*)d_in[18];
  const float* cxW   = (const float*)d_in[19];
  const float* cxb   = (const float*)d_in[20];
  const float* chW   = (const float*)d_in[21];
  const float* chb   = (const float*)d_in[22];
  const float* wc    = (const float*)d_in[23];
  const float* gb    = (const float*)d_in[24];
  const float* linW  = (const float*)d_in[25];
  const float* linb  = (const float*)d_in[26];
  float* out = (float*)d_out;

  // ---- workspace carve-up (all 256B aligned)
  char* ws = (char*)d_ws;
  size_t off = 0;
  auto take = [&](size_t bytes) -> char* {
    char* p = ws + off;
    off = (off + bytes + 255) & ~(size_t)255;
    return p;
  };
  int*   csrSrc = (int*)  take((size_t)NB*EE*4);
  float* csrW   = (float*)take((size_t)NB*EE*4);
  int*   csrCnt = (int*)  take(NB*4);
  f16*   Xpad   = (f16*)  take((size_t)ROWS*32*2);
  f16*   W1p    = (f16*)  take((size_t)8192*2);
  f16*   W2p    = (f16*)  take((size_t)65536*2);
  f16*   W3p    = (f16*)  take((size_t)8192*2);
  f16*   nW1p   = (f16*)  take((size_t)128*8192*2);
  f16*   nW2p   = (f16*)  take((size_t)128*65536*2);
  f16*   WcatP  = (f16*)  take((size_t)2*1572864*2);
  float* biasc  = (float*)take(2048*4);
  float* scaleB = (float*)take(256*4);
  float* shiftB = (float*)take(256*4);
  float* Yb     = (float*)take((size_t)ROWS*256*4);   // reused f32 GEMM out
  f16*   Hb     = (f16*)  take((size_t)ROWS*256*2);   // reused f16 activations
  f16*   nodeA  = (f16*)  take((size_t)128*256*32*2);
  f16*   embT   = (f16*)  take((size_t)TT*1024*256*2);
  f16*   Acat   = (f16*)  take((size_t)1024*1536*2);
  float* Gates  = (float*)take((size_t)1024*1024*4);
  f16*   Hs     = (f16*)  take((size_t)2*1024*256*2);
  float* Cs     = (float*)take((size_t)2*1024*256*4);
  (void)in_sizes; (void)n_in; (void)out_size; (void)ws_size;

  auto gemm = [&](const f16* A, const f16* Bp, float* C, const float* bias,
                  int M, int N, int K, int lda, int ldc,
                  long sAb, long sBb, long sCb, long sBiasB, int batch) {
    int tiles = (M/32) * (N/32);
    dim3 g((tiles + 7) / 8, batch);
    k_wmma_gemm<<<g, 256, 0, stream>>>(A, Bp, C, bias, M, N, K, lda, ldc,
                                       sAb, sBb, sCb, sBiasB);
  };

  // ---- graph + packs + init
  k_graph<<<1, 128, 0, stream>>>(e_i, e_a, csrSrc, csrW, csrCnt);
  k_prep_x<<<(ROWS*32)/256, 256, 0, stream>>>(input, Xpad);
  k_pack_b<<<dim3(32, 1),   256, 0, stream>>>(W1, W1p, 16, 256, 32, 256, 0, 0);
  k_pack_b<<<dim3(256, 1),  256, 0, stream>>>(W2, W2p, 256, 256, 256, 256, 0, 0);
  k_pack_b<<<dim3(32, 1),   256, 0, stream>>>(W3, W3p, 256, 8, 256, 32, 0, 0);
  k_pack_b<<<dim3(32, 128), 256, 0, stream>>>(nW1, nW1p, 8, 256, 32, 256, 8L*256, 8192);
  k_pack_b<<<dim3(256,128), 256, 0, stream>>>(nW2, nW2p, 256, 256, 256, 256, 65536L, 65536L);
  k_pack_wcat<<<dim3(6144, 2), 256, 0, stream>>>(cxW, chW, WcatP);
  k_biascat<<<8, 256, 0, stream>>>(cxb, chb, gb, biasc);
  k_zero<<<1024, 256, 0, stream>>>((unsigned int*)Hs, 2L*1024*256*2/4);
  k_zero<<<2048, 256, 0, stream>>>((unsigned int*)Cs, 2L*1024*256);

  // ---- nn_model2 MLP (BN makes b1/b2/b3 cancel -> no GEMM bias)
  gemm(Xpad, W1p, Yb, nullptr, ROWS, 256, 32, 32, 256, 0, 0, 0, 0, 1);
  k_colstats<<<256, 256, 0, stream>>>(Yb, ROWS, 256, bn1g, bn1b, scaleB, shiftB);
  k_bnact<<<(ROWS*256)/256, 256, 0, stream>>>(Yb, Hb, scaleB, shiftB, 256, (long)ROWS*256, 1);

  gemm(Hb, W2p, Yb, nullptr, ROWS, 256, 256, 256, 256, 0, 0, 0, 0, 1);
  k_colstats<<<256, 256, 0, stream>>>(Yb, ROWS, 256, bn2g, bn2b, scaleB, shiftB);
  k_bnact<<<(ROWS*256)/256, 256, 0, stream>>>(Yb, Hb, scaleB, shiftB, 256, (long)ROWS*256, 1);

  gemm(Hb, W3p, Yb, nullptr, ROWS, 32, 256, 256, 32, 0, 0, 0, 0, 1);
  k_colstats<<<8, 256, 0, stream>>>(Yb, ROWS, 32, bn3g, bn3b, scaleB, shiftB);
  k_bn3_reorder<<<(128*256*32)/256, 256, 0, stream>>>(Yb, nodeA, scaleB, shiftB);

  // ---- per-node MLPs (batched over 128 nodes)
  gemm(nodeA, nW1p, Yb, nb1, 256, 256, 32, 32, 256, 8192, 8192, 65536, 256, 128);
  k_cvt<<<(128*65536)/256, 256, 0, stream>>>(Yb, Hb, 128L*65536, 1);
  gemm(Hb, nW2p, Yb, nb2, 256, 256, 256, 256, 256, 65536, 65536, 65536, 256, 128);
  k_emb<<<(128*65536)/256, 256, 0, stream>>>(Yb, embT);

  // ---- ChebConv-LSTM: T=32 steps x L=2 layers, one fused GEMM per cell
  for (int t = 0; t < TT; ++t) {
    for (int l = 0; l < 2; ++l) {
      const f16* xsrc = (l == 0) ? (embT + (long)t*1024*256) : Hs;  // layer-0 h already updated
      k_copy16<<<1024, 256, 0, stream>>>(Acat,       1536, xsrc,                256);
      k_copy16<<<1024, 256, 0, stream>>>(Acat + 768, 1536, Hs + (long)l*262144, 256);
      k_prop<<<dim3(1024, 2), 256, 0, stream>>>(Acat, csrSrc, csrW, csrCnt, 0); // Px, Ph
      k_prop<<<dim3(1024, 2), 256, 0, stream>>>(Acat, csrSrc, csrW, csrCnt, 1); // PPx, PPh
      gemm(Acat, WcatP + (long)l*1572864, Gates, biasc + l*1024,
           1024, 1024, 1536, 1536, 1024, 0, 0, 0, 0, 1);
      k_gates<<<1024, 256, 0, stream>>>(Gates, wc + l*768,
                                        Cs + (long)l*262144, Hs + (long)l*262144);
    }
    k_out<<<1024, 256, 0, stream>>>(Hs + 262144, linW, linb, out, t);
  }
}
